// OffsetAttentionLayer_32435593020119
// MI455X (gfx1250) — matrix operand order repaired and checked by reading, compiled, not run on hardware
//
#include <hip/hip_runtime.h>
#include <stdint.h>

#define BATCHES 8
#define LMAXC   2560
#define DIMD    256
#define PADROWS (BATCHES * LMAXC)   // 20480
#define NQT     (LMAXC / 16)        // 160 query tiles per batch

typedef __bf16 bf16_t;
typedef __attribute__((ext_vector_type(8)))  __bf16 v8bf;
typedef __attribute__((ext_vector_type(16))) __bf16 v16bf;
typedef __attribute__((ext_vector_type(8)))  float  v8f;

// ---------- helpers ----------

__device__ __forceinline__ bf16_t f2bf(float f) {
  unsigned u = __builtin_bit_cast(unsigned, f);
  unsigned r = (u + 0x7FFFu + ((u >> 16) & 1u)) >> 16;
  unsigned short s = (unsigned short)r;
  return __builtin_bit_cast(bf16_t, s);
}

// A-operand (16xK=32 bf16), ISA layout: lanes 0-15 hold row m=lane with
// K chunks {0..7},{16..23}; lanes 16-31 hold row m=lane-16 with K chunks
// {8..15},{24..31}.  base points at (row0, k0); lda in elements.
__device__ __forceinline__ v16bf load_a16(const bf16_t* base, int lda, int lane) {
  int m  = lane & 15;
  int hi = lane >> 4;
  const bf16_t* p = base + (size_t)m * lda + hi * 8;
  v8bf c0 = *(const v8bf*)(p);
  v8bf c1 = *(const v8bf*)(p + 16);
  v16bf a;
#pragma unroll
  for (int i = 0; i < 8; ++i) { a[i] = c0[i]; a[i + 8] = c1[i]; }
  return a;
}

// B-operand (K=32 x 16 bf16) where B[k][n] = W[n][k] with W row-major,
// stride ldb.  ISA layout: lanes 0-15 = col n, K chunks {0..7},{16..23};
// lanes 16-31 = col n, K chunks {8..15},{24..31}.  base points at (col0 row of W, k0).
__device__ __forceinline__ v16bf load_b16(const bf16_t* base, int ldb, int lane) {
  int n  = lane & 15;
  int hi = lane >> 4;
  const bf16_t* p = base + (size_t)n * ldb + hi * 8;
  v8bf c0 = *(const v8bf*)(p);
  v8bf c1 = *(const v8bf*)(p + 16);
  v16bf b;
#pragma unroll
  for (int i = 0; i < 8; ++i) { b[i] = c0[i]; b[i + 8] = c1[i]; }
  return b;
}

__device__ __forceinline__ v8f wmma_bf16(v16bf a, v16bf b, v8f c) {
  return __builtin_amdgcn_wmma_f32_16x16x32_bf16(false, a, false, b,
                                                 (short)0, c, false, false);
}

// ---------- kernel 0: counts / offsets ----------

__global__ void counts_kernel(const int* __restrict__ batch_ids, int N,
                              int* __restrict__ counts, int* __restrict__ offsets) {
  __shared__ int c[BATCHES];
  if (threadIdx.x < BATCHES) c[threadIdx.x] = 0;
  __syncthreads();
  for (int n = threadIdx.x; n < N; n += blockDim.x)
    atomicAdd(&c[batch_ids[n]], 1);
  __syncthreads();
  if (threadIdx.x == 0) {
    int off = 0;
    for (int b = 0; b < BATCHES; ++b) { counts[b] = c[b]; offsets[b] = off; off += c[b]; }
  }
}

// ---------- kernel 1: weight conversion f32 -> bf16 ----------

__global__ void convert_w_kernel(const float* __restrict__ in_w,
                                 const float* __restrict__ out_w,
                                 bf16_t* __restrict__ in_wb,
                                 bf16_t* __restrict__ out_wb) {
  int i = blockIdx.x * 256 + threadIdx.x;
  if (i < 3 * DIMD * DIMD) in_wb[i] = f2bf(in_w[i]);
  if (i < DIMD * DIMD)     out_wb[i] = f2bf(out_w[i]);
}

// ---------- kernel 2: posenc + residual + padded bf16 pack ----------

__global__ __launch_bounds__(256)
void posenc_pad_kernel(const float* __restrict__ features,
                       const float* __restrict__ coords,
                       const float* __restrict__ Wp, const float* __restrict__ bp,
                       const int* __restrict__ counts, const int* __restrict__ offsets,
                       float* __restrict__ resid, bf16_t* __restrict__ padded) {
  int row = blockIdx.x;            // 0..PADROWS-1
  int d   = threadIdx.x;           // 0..255
  int b = row / LMAXC, l = row % LMAXC;
  float val = 0.f;
  if (l < counts[b]) {
    int n = offsets[b] + l;
    float v = features[(size_t)n * DIMD + d]
            + coords[2 * n]     * Wp[2 * d]
            + coords[2 * n + 1] * Wp[2 * d + 1]
            + bp[d];
    resid[(size_t)n * DIMD + d] = v;
    val = v;
  }
  padded[(size_t)row * DIMD + d] = f2bf(val);
}

// ---------- kernel 3: QKV projection GEMM (WMMA bf16) ----------
// C[20480,768] = padded @ in_w^T + in_b ; q scaled by 1/16; v stored
// transposed vt[b][d][l].

__global__ __launch_bounds__(128)
void qkv_gemm_kernel(const bf16_t* __restrict__ A, const bf16_t* __restrict__ W,
                     const float* __restrict__ in_b,
                     bf16_t* __restrict__ qb, bf16_t* __restrict__ kb,
                     bf16_t* __restrict__ vt) {
  const int NT64 = (3 * DIMD) / 64;            // 12
  int wave = threadIdx.x >> 5, lane = threadIdx.x & 31;
  int gw   = blockIdx.x * 4 + wave;            // 15360 waves
  int mt   = gw / NT64;
  int nt64 = gw % NT64;
  int hi = lane >> 4, nn = lane & 15;

  const bf16_t* arow = A + (size_t)(mt * 16) * DIMD;
  v8f acc[4];
#pragma unroll
  for (int t = 0; t < 4; ++t) acc[t] = v8f{};

#pragma unroll
  for (int c = 0; c < 8; ++c) {                 // K = 256 in chunks of 32
    v16bf a = load_a16(arow + c * 32, DIMD, lane);
#pragma unroll
    for (int t = 0; t < 4; ++t) {
      int n0 = nt64 * 64 + t * 16;
      v16bf bm = load_b16(W + (size_t)n0 * DIMD + c * 32, DIMD, lane);
      acc[t] = wmma_bf16(a, bm, acc[t]);
    }
  }

  int row0 = mt * 16;
  int b  = row0 / LMAXC;                        // 16-row tile never crosses batches
  int l0 = row0 % LMAXC;
#pragma unroll
  for (int t = 0; t < 4; ++t) {
    int ng = nt64 * 64 + t * 16 + nn;
    float bias = in_b[ng];
#pragma unroll
    for (int i = 0; i < 8; ++i) {
      int M = i + 8 * hi;
      size_t row = (size_t)(row0 + M);
      float v = acc[t][i] + bias;
      if (ng < DIMD) {
        qb[row * DIMD + ng] = f2bf(v * 0.0625f);            // 1/sqrt(256)
      } else if (ng < 2 * DIMD) {
        kb[row * DIMD + (ng - DIMD)] = f2bf(v);
      } else {
        vt[((size_t)b * DIMD + (ng - 2 * DIMD)) * LMAXC + (l0 + M)] = f2bf(v);
      }
    }
  }
}

// ---------- kernel 4: flash attention (WMMA bf16, online softmax) ----------

__global__ __launch_bounds__(128)
void attn_kernel(const bf16_t* __restrict__ qb, const bf16_t* __restrict__ kb,
                 const bf16_t* __restrict__ vt, bf16_t* __restrict__ ctxb,
                 const int* __restrict__ counts) {
  __shared__ __align__(16) bf16_t Pbuf[4][16 * 32];
  int wave = threadIdx.x >> 5, lane = threadIdx.x & 31;
  int gw = blockIdx.x * 4 + wave;               // 1280 waves total
  int b  = gw / NQT;
  int qt = gw % NQT;
  int count = counts[b];
  if (qt * 16 >= count) return;                 // wave-uniform

  int hi = lane >> 4, nn = lane & 15;
  const bf16_t* qrow = qb + ((size_t)b * LMAXC + qt * 16) * DIMD;

  v16bf Aq[8];
#pragma unroll
  for (int c = 0; c < 8; ++c) Aq[c] = load_a16(qrow + c * 32, DIMD, lane);

  v8f ctx[16];
#pragma unroll
  for (int t = 0; t < 16; ++t) ctx[t] = v8f{};
  float mrow[8], lrow[8];
#pragma unroll
  for (int i = 0; i < 8; ++i) { mrow[i] = -1e30f; lrow[i] = 0.f; }

  int nkt = (count + 31) / 32;
  for (int kt = 0; kt < nkt; ++kt) {
    int kb0 = kt * 32;
    const bf16_t* krow = kb + ((size_t)b * LMAXC + kb0) * DIMD;
    v8f S0 = v8f{}, S1 = v8f{};
#pragma unroll
    for (int c = 0; c < 8; ++c) {
      v16bf Bk0 = load_b16(krow + c * 32, DIMD, lane);
      v16bf Bk1 = load_b16(krow + (size_t)16 * DIMD + c * 32, DIMD, lane);
      S0 = wmma_bf16(Aq[c], Bk0, S0);
      S1 = wmma_bf16(Aq[c], Bk1, S1);
    }
    // mask tail keys (key index = kb0 + nn [+16])
    if (kb0 + 32 > count) {
      if (kb0 + nn >= count) {
#pragma unroll
        for (int i = 0; i < 8; ++i) S0[i] = -1e9f;
      }
      if (kb0 + 16 + nn >= count) {
#pragma unroll
        for (int i = 0; i < 8; ++i) S1[i] = -1e9f;
      }
    }
    // online softmax: row r = VGPR i (+8 if hi); reduce across 16-lane half
    float rmax[8];
#pragma unroll
    for (int i = 0; i < 8; ++i) rmax[i] = fmaxf(S0[i], S1[i]);
#pragma unroll
    for (int msk = 1; msk < 16; msk <<= 1)
#pragma unroll
      for (int i = 0; i < 8; ++i)
        rmax[i] = fmaxf(rmax[i], __shfl_xor(rmax[i], msk, 32));

    float alpha[8];
#pragma unroll
    for (int i = 0; i < 8; ++i) {
      float mn = fmaxf(mrow[i], rmax[i]);
      alpha[i] = __expf(mrow[i] - mn);
      mrow[i] = mn;
    }
    float rsum[8];
#pragma unroll
    for (int i = 0; i < 8; ++i) {
      float p0 = __expf(S0[i] - mrow[i]);
      float p1 = __expf(S1[i] - mrow[i]);
      rsum[i] = p0 + p1;
      Pbuf[wave][(i + 8 * hi) * 32 + nn]      = f2bf(p0);
      Pbuf[wave][(i + 8 * hi) * 32 + 16 + nn] = f2bf(p1);
    }
#pragma unroll
    for (int msk = 1; msk < 16; msk <<= 1)
#pragma unroll
      for (int i = 0; i < 8; ++i)
        rsum[i] += __shfl_xor(rsum[i], msk, 32);
#pragma unroll
    for (int i = 0; i < 8; ++i) lrow[i] = lrow[i] * alpha[i] + rsum[i];
#pragma unroll
    for (int t = 0; t < 16; ++t)
#pragma unroll
      for (int i = 0; i < 8; ++i) ctx[t][i] *= alpha[i];

    // order LDS stores before A-layout reload (same-wave DS is in-order)
    asm volatile("s_wait_dscnt 0" ::: "memory");
    v16bf Ap = load_a16(&Pbuf[wave][0], 32, lane);

    const bf16_t* vrow = vt + (size_t)b * DIMD * LMAXC + kb0;
#pragma unroll
    for (int t = 0; t < 16; ++t) {
      v16bf Bv = load_b16(vrow + (size_t)(t * 16) * LMAXC, LMAXC, lane);
      ctx[t] = wmma_bf16(Ap, Bv, ctx[t]);
    }
  }

  float inv[8];
#pragma unroll
  for (int i = 0; i < 8; ++i) inv[i] = 1.0f / lrow[i];
  bf16_t* crow = ctxb + ((size_t)b * LMAXC + qt * 16) * DIMD;
#pragma unroll
  for (int t = 0; t < 16; ++t)
#pragma unroll
    for (int i = 0; i < 8; ++i) {
      int M = i + 8 * hi;
      crow[(size_t)M * DIMD + t * 16 + nn] = f2bf(ctx[t][i] * inv[i]);
    }
}

// ---------- kernel 5: output projection GEMM (WMMA bf16) ----------

__global__ __launch_bounds__(128)
void out_gemm_kernel(const bf16_t* __restrict__ A, const bf16_t* __restrict__ W,
                     const float* __restrict__ out_b, float* __restrict__ attn_out) {
  const int NT64 = DIMD / 64;                  // 4
  int wave = threadIdx.x >> 5, lane = threadIdx.x & 31;
  int gw   = blockIdx.x * 4 + wave;            // 5120 waves
  int mt   = gw / NT64;
  int nt64 = gw % NT64;
  int hi = lane >> 4, nn = lane & 15;

  const bf16_t* arow = A + (size_t)(mt * 16) * DIMD;
  v8f acc[4];
#pragma unroll
  for (int t = 0; t < 4; ++t) acc[t] = v8f{};

#pragma unroll
  for (int c = 0; c < 8; ++c) {
    v16bf a = load_a16(arow + c * 32, DIMD, lane);
#pragma unroll
    for (int t = 0; t < 4; ++t) {
      int n0 = nt64 * 64 + t * 16;
      v16bf bm = load_b16(W + (size_t)n0 * DIMD + c * 32, DIMD, lane);
      acc[t] = wmma_bf16(a, bm, acc[t]);
    }
  }

  int row0 = mt * 16;
#pragma unroll
  for (int t = 0; t < 4; ++t) {
    int ng = nt64 * 64 + t * 16 + nn;
    float bias = out_b[ng];
#pragma unroll
    for (int i = 0; i < 8; ++i) {
      int M = i + 8 * hi;
      attn_out[(size_t)(row0 + M) * DIMD + ng] = acc[t][i] + bias;
    }
  }
}

// ---------- kernel 6: gather + LayerNorm + residual + ReLU ----------

__global__ __launch_bounds__(256)
void ln_kernel(const float* __restrict__ attn_out, const float* __restrict__ resid,
               const float* __restrict__ gamma, const float* __restrict__ beta,
               const int* __restrict__ batch_ids, const int* __restrict__ offsets,
               float* __restrict__ out) {
  __shared__ float red[256];
  int n = blockIdx.x, d = threadIdx.x;
  int b = batch_ids[n];
  int l = n - offsets[b];
  float x = attn_out[((size_t)b * LMAXC + l) * DIMD + d];

  red[d] = x;
  __syncthreads();
  for (int s = 128; s > 0; s >>= 1) {
    if (d < s) red[d] += red[d + s];
    __syncthreads();
  }
  float mu = red[0] * (1.0f / DIMD);
  __syncthreads();
  float dx = x - mu;
  red[d] = dx * dx;
  __syncthreads();
  for (int s = 128; s > 0; s >>= 1) {
    if (d < s) red[d] += red[d + s];
    __syncthreads();
  }
  float var = red[0] * (1.0f / DIMD);
  float ln = dx * rsqrtf(var + 1e-6f) * gamma[d] + beta[d];
  float r = ln + resid[(size_t)n * DIMD + d];
  out[(size_t)n * DIMD + d] = fmaxf(r, 0.f);
}

// ---------- launcher ----------

extern "C" void kernel_launch(void* const* d_in, const int* in_sizes, int n_in,
                              void* d_out, int out_size, void* d_ws, size_t ws_size,
                              hipStream_t stream) {
  const float* features = (const float*)d_in[0];
  const float* coords   = (const float*)d_in[1];
  const float* Wp       = (const float*)d_in[2];
  const float* bp       = (const float*)d_in[3];
  const float* in_w     = (const float*)d_in[4];
  const float* in_b     = (const float*)d_in[5];
  const float* out_w    = (const float*)d_in[6];
  const float* out_b    = (const float*)d_in[7];
  const float* gamma    = (const float*)d_in[8];
  const float* beta     = (const float*)d_in[9];
  const int*   batch_ids = (const int*)d_in[10];
  int N = in_sizes[10];

  char* ws = (char*)d_ws;
  size_t off = 0;
  auto alloc = [&](size_t bytes) {
    void* p = ws + off;
    off += (bytes + 255) & ~(size_t)255;
    return p;
  };
  int*    counts   = (int*)alloc(BATCHES * sizeof(int));
  int*    offsets  = (int*)alloc(BATCHES * sizeof(int));
  float*  resid    = (float*)alloc((size_t)N * DIMD * sizeof(float));
  bf16_t* padded   = (bf16_t*)alloc((size_t)PADROWS * DIMD * sizeof(bf16_t));
  bf16_t* qb       = (bf16_t*)alloc((size_t)PADROWS * DIMD * sizeof(bf16_t));
  bf16_t* kb       = (bf16_t*)alloc((size_t)PADROWS * DIMD * sizeof(bf16_t));
  bf16_t* vt       = (bf16_t*)alloc((size_t)PADROWS * DIMD * sizeof(bf16_t));
  bf16_t* ctxb     = (bf16_t*)alloc((size_t)PADROWS * DIMD * sizeof(bf16_t));
  float*  attn_out = (float*)alloc((size_t)PADROWS * DIMD * sizeof(float));
  bf16_t* in_wb    = (bf16_t*)alloc((size_t)3 * DIMD * DIMD * sizeof(bf16_t));
  bf16_t* out_wb   = (bf16_t*)alloc((size_t)DIMD * DIMD * sizeof(bf16_t));
  (void)ws_size;

  counts_kernel<<<1, 256, 0, stream>>>(batch_ids, N, counts, offsets);
  convert_w_kernel<<<(3 * DIMD * DIMD + 255) / 256, 256, 0, stream>>>(in_w, out_w, in_wb, out_wb);
  posenc_pad_kernel<<<PADROWS, 256, 0, stream>>>(features, coords, Wp, bp,
                                                 counts, offsets, resid, padded);
  {
    int waves = (PADROWS / 16) * ((3 * DIMD) / 64);   // 15360
    qkv_gemm_kernel<<<waves / 4, 128, 0, stream>>>(padded, in_wb, in_b, qb, kb, vt);
  }
  {
    int waves = BATCHES * NQT;                         // 1280
    attn_kernel<<<waves / 4, 128, 0, stream>>>(qb, kb, vt, ctxb, counts);
  }
  {
    int waves = (PADROWS / 16) * (DIMD / 64);          // 5120
    out_gemm_kernel<<<waves / 4, 128, 0, stream>>>(ctxb, out_wb, out_b, attn_out);
  }
  ln_kernel<<<N, 256, 0, stream>>>(attn_out, resid, gamma, beta, batch_ids, offsets,
                                   (float*)d_out);
}
